// LNNRegression_83262236000493
// MI455X (gfx1250) — matrix operand order repaired and checked by reading, not compiled, not run on hardware
//
#include <hip/hip_runtime.h>
#include <math.h>

#define B_TOT 1024
#define T_LEN 512
#define D_IN  31
#define NTILE (B_TOT/16)
#define TCH   64

typedef __attribute__((ext_vector_type(16))) __bf16 v16bf;
typedef __attribute__((ext_vector_type(8)))  float  v8f;

#define LOG2E 1.4426950408889634f

__device__ __forceinline__ v8f vzero8() {
  v8f c = {0.f,0.f,0.f,0.f,0.f,0.f,0.f,0.f};
  return c;
}

__device__ __forceinline__ v16bf bfzero16() {
  v16bf v;
#pragma unroll
  for (int e = 0; e < 16; ++e) v[e] = (__bf16)0.f;
  return v;
}

// Branch-free tanh: hardware V_TANH_F32 if the builtin exists, else exp2+rcp.
__device__ __forceinline__ float fast_tanh(float x) {
#if __has_builtin(__builtin_amdgcn_tanhf)
  return __builtin_amdgcn_tanhf(x);
#else
  float ax = fabsf(x);
  float e  = __builtin_amdgcn_exp2f(ax * (-2.f * LOG2E));
  float r  = (1.f - e) * __builtin_amdgcn_rcpf(1.f + e);
  return copysignf(r, x);
#endif
}

// Branch-free sigmoid.
__device__ __forceinline__ float fast_sigmoid(float a) {
  return __builtin_amdgcn_rcpf(1.f + __builtin_amdgcn_exp2f(-a * LOG2E));
}

__device__ __forceinline__ float fast_exp(float a) {
  return __builtin_amdgcn_exp2f(a * LOG2E);
}

__device__ __forceinline__ v8f wmma_bf16(v16bf a, v16bf b, v8f c) {
  // D = A(16x32 bf16) * B(32x16 bf16) + C(16x16 f32)
  return __builtin_amdgcn_wmma_f32_16x16x32_bf16(false, a, false, b, (short)0, c, false, false);
}

// Build B-fragment (32x16, bf16) for W row-major [N x K]: column n of B = row n of W.
// lane n0 (0..15) holds K = 16*h .. 16*h+15 of column (n), h = lane>>4.
__device__ __forceinline__ v16bf load_bfrag(const float* __restrict__ W, int ldw,
                                            int n, int kbase, int kmax) {
  v16bf r;
#pragma unroll
  for (int e = 0; e < 16; ++e) {
    int k = kbase + e;
    r[e] = (__bf16)((k < kmax) ? W[n * ldw + k] : 0.f);
  }
  return r;
}

// A-fragment (16x32 bf16) gathered from LDS tile sh[16][68] (fp32), K window at kofs.
// lane: row = lane&15; K=kofs+16h+e
__device__ __forceinline__ v16bf afrag_lds(const float* __restrict__ sh, int kofs, int lane) {
  const int row = lane & 15, h = lane >> 4;
  const float* p = sh + row * 68 + kofs + 16 * h;
  v16bf r;
#pragma unroll
  for (int e = 0; e < 8; ++e) r[e] = (__bf16)p[e];
#pragma unroll
  for (int e = 0; e < 8; ++e) r[8 + e] = (__bf16)p[8 + e];
  return r;
}

// Load x row in A-fragment per-lane element order (K pattern), zero-pad K>=31.
__device__ __forceinline__ void load_xpat(const float* __restrict__ x, int brow, int t,
                                          int h, float o[16]) {
  const float* p = x + ((size_t)brow * T_LEN + t) * D_IN;
#pragma unroll
  for (int e = 0; e < 8; ++e) { int k = 16*h + e; o[e] = p[k]; }         // k <= 23, always valid
#pragma unroll
  for (int e = 0; e < 8; ++e) { int k = 16*h + 8 + e; o[8+e] = (k < D_IN) ? p[k] : 0.f; }
}

// ---------------------------------------------------------------------------
// Kernel 1: FlashFloodGate (WMMA) + repack x into bf16 A-fragments.
// grid (64 tiles, 8 t-chunks), 1 wave/block.
// ---------------------------------------------------------------------------
__global__ __launch_bounds__(32) void k1_gate_repack(
    const float* __restrict__ x,
    const float* __restrict__ fg_w1, const float* __restrict__ fg_b1,
    const float* __restrict__ fg_w2, const float* __restrict__ fg_b2,
    unsigned short* __restrict__ fragx, float* __restrict__ invtau) {
  const int lane = threadIdx.x, n0 = lane & 15, h = lane >> 4;
  const int tile = blockIdx.x;
  const int t0 = blockIdx.y * TCH;
  const int brow = tile * 16 + n0;

  // gate layer-1 weight B-fragments; ci padded [x(31),0, dx(31),0] -> K=64
  v16bf g1B[2][2];
#pragma unroll
  for (int nt = 0; nt < 2; ++nt)
#pragma unroll
    for (int kt = 0; kt < 2; ++kt) {
      v16bf r;
      const int n = nt * 16 + n0;
#pragma unroll
      for (int e = 0; e < 16; ++e) {
        int k = kt * 32 + 16 * h + e;       // padded K index 0..63
        float w = 0.f;
        if (k < 31)                 w = fg_w1[n * 62 + k];        // x part
        else if (k >= 32 && k < 63) w = fg_w1[n * 62 + (k - 1)];  // dx part
        r[e] = (__bf16)w;
      }
      g1B[nt][kt] = r;
    }
  const float b1a = fg_b1[n0], b1b = fg_b1[16 + n0];
  const float w2a = fg_w2[n0], w2b = fg_w2[16 + n0];
  const float b2  = fg_b2[0];

  float xp[16], xc[16];
  load_xpat(x, brow, (t0 > 0) ? (t0 - 1) : 0, h, xp);

  for (int t = t0; t < t0 + TCH; ++t) {
    load_xpat(x, brow, t, h, xc);
    v16bf ax, adx;
#pragma unroll
    for (int e = 0; e < 16; ++e) {
      ax[e]  = (__bf16)xc[e];
      adx[e] = (__bf16)(xc[e] - xp[e]);
    }
    ((v16bf*)fragx)[((size_t)tile * T_LEN + t) * 32 + lane] = ax;

    v8f c0 = vzero8(), c1 = vzero8();
    c0 = wmma_bf16(ax,  g1B[0][0], c0);
    c0 = wmma_bf16(adx, g1B[0][1], c0);
    c1 = wmma_bf16(ax,  g1B[1][0], c1);
    c1 = wmma_bf16(adx, g1B[1][1], c1);

    float pr[8];
#pragma unroll
    for (int r = 0; r < 8; ++r)
      pr[r] = fast_tanh(c0[r] + b1a) * w2a + fast_tanh(c1[r] + b1b) * w2b;
#pragma unroll
    for (int r = 0; r < 8; ++r)
#pragma unroll
      for (int m = 1; m < 16; m <<= 1)
        pr[r] += __shfl_xor(pr[r], m, 16);

    if (n0 == 0) {
#pragma unroll
      for (int r = 0; r < 8; ++r) {
        float cs  = fast_sigmoid(pr[r] + b2);
        float tau = 10.0f - cs * 9.99f;
        invtau[((size_t)tile * T_LEN + t) * 16 + 8 * h + r] = 1.0f / tau;
      }
    }
#pragma unroll
    for (int e = 0; e < 16; ++e) xp[e] = xc[e];
  }
}

// ---------------------------------------------------------------------------
// Kernel 2: fused LTC scan + fusion + attention (online softmax).
// One wave per 16-batch tile; weights live in VGPR B-fragments.
// ---------------------------------------------------------------------------
__global__ __launch_bounds__(32) void k2_scan(
    const unsigned short* __restrict__ fragx, const float* __restrict__ invtau,
    const float* __restrict__ fwin, const float* __restrict__ fbin, const float* __restrict__ fwrec,
    const float* __restrict__ swin, const float* __restrict__ sbin, const float* __restrict__ swrec,
    const float* __restrict__ fu_w, const float* __restrict__ fu_b,
    const float* __restrict__ at_w1, const float* __restrict__ at_b1,
    const float* __restrict__ at_w2, const float* __restrict__ at_b2,
    float* __restrict__ ctxg) {
  __shared__ float shA[16 * 68];   // h transpose buffer (fast cols 0-31, slow 32-63)
  __shared__ float shB[16 * 68];   // ltc transpose buffer
  const int lane = threadIdx.x, n0 = lane & 15, h = lane >> 4;
  const int tile = blockIdx.x;

  v16bf fwinB[2], fwrecB[2], swinB[2], swrecB[2], fuB[4][2], at1B[2][2];
#pragma unroll
  for (int nt = 0; nt < 2; ++nt) {
    fwinB[nt]  = load_bfrag(fwin,  31, nt*16 + n0, 16*h, 31);
    fwrecB[nt] = load_bfrag(fwrec, 32, nt*16 + n0, 16*h, 32);
    swinB[nt]  = load_bfrag(swin,  31, nt*16 + n0, 16*h, 31);
    swrecB[nt] = load_bfrag(swrec, 32, nt*16 + n0, 16*h, 32);
  }
#pragma unroll
  for (int nt = 0; nt < 4; ++nt)
#pragma unroll
    for (int kt = 0; kt < 2; ++kt)
      fuB[nt][kt] = load_bfrag(fu_w, 64, nt*16 + n0, kt*32 + 16*h, 64);
#pragma unroll
  for (int nt = 0; nt < 2; ++nt)
#pragma unroll
    for (int kt = 0; kt < 2; ++kt)
      at1B[nt][kt] = load_bfrag(at_w1, 64, nt*16 + n0, kt*32 + 16*h, 64);

  const float fbF[2] = { fbin[n0], fbin[16 + n0] };
  const float fbS[2] = { sbin[n0], sbin[16 + n0] };
  float fub[4];
#pragma unroll
  for (int nt = 0; nt < 4; ++nt) fub[nt] = fu_b[nt*16 + n0];
  const float ab1[2] = { at_b1[n0], at_b1[16 + n0] };
  const float w2a = at_w2[n0], w2b = at_w2[16 + n0];
  const float ab2 = at_b2[0];

  v8f hf[2] = { vzero8(), vzero8() }, hs[2] = { vzero8(), vzero8() };
  v16bf hAf = bfzero16(), hAs = bfzero16();
  v8f ctx[4] = { vzero8(), vzero8(), vzero8(), vzero8() };
  float mrow[8], srow[8];
#pragma unroll
  for (int r = 0; r < 8; ++r) { mrow[r] = -3.0e38f; srow[r] = 0.f; }

  const v16bf* fx = (const v16bf*)fragx;
  const size_t fbase = (size_t)tile * T_LEN * 32 + lane;
  v16bf ax = fx[fbase];
  float inv8[8];
  {
    const float* tp = invtau + (size_t)tile * T_LEN * 16 + 8 * h;
#pragma unroll
    for (int r = 0; r < 8; ++r) inv8[r] = tp[r];
  }

  for (int t = 0; t < T_LEN; ++t) {
    // software-pipeline next-step inputs (and prefetch a few steps ahead)
    v16bf axn = ax;
    float invn[8];
#pragma unroll
    for (int r = 0; r < 8; ++r) invn[r] = inv8[r];
    if (t + 1 < T_LEN) {
      axn = fx[fbase + (size_t)(t + 1) * 32];
      const float* tp = invtau + ((size_t)tile * T_LEN + (t + 1)) * 16 + 8 * h;
#pragma unroll
      for (int r = 0; r < 8; ++r) invn[r] = tp[r];
    }
    if (t + 8 < T_LEN) __builtin_prefetch(fx + fbase + (size_t)(t + 8) * 32, 0, 1);

    // --- fast LTC cell: pre = x*fwin^T ; rec = h*fwrec^T ---
#pragma unroll
    for (int nt = 0; nt < 2; ++nt) {
      v8f acc = vzero8();
      acc = wmma_bf16(ax,  fwinB[nt],  acc);
      acc = wmma_bf16(hAf, fwrecB[nt], acc);
#pragma unroll
      for (int r = 0; r < 8; ++r) {
        float u = fast_tanh(acc[r] + fbF[nt]);
        hf[nt][r] += (u - hf[nt][r]) * inv8[r];
      }
    }
    // --- slow LTC cell (tau = 5.0) ---
#pragma unroll
    for (int nt = 0; nt < 2; ++nt) {
      v8f acc = vzero8();
      acc = wmma_bf16(ax,  swinB[nt],  acc);
      acc = wmma_bf16(hAs, swrecB[nt], acc);
#pragma unroll
      for (int r = 0; r < 8; ++r) {
        float u = fast_tanh(acc[r] + fbS[nt]);
        hs[nt][r] += (u - hs[nt][r]) * 0.2f;
      }
    }
    // --- transpose h (C layout -> A layout) via LDS ---
#pragma unroll
    for (int nt = 0; nt < 2; ++nt)
#pragma unroll
      for (int r = 0; r < 8; ++r) {
        shA[(r + 8*h) * 68 + nt*16 + n0]      = hf[nt][r];
        shA[(r + 8*h) * 68 + 32 + nt*16 + n0] = hs[nt][r];
      }
    __syncthreads();
    hAf = afrag_lds(shA, 0,  lane);
    hAs = afrag_lds(shA, 32, lane);

    // --- fusion: ltc = tanh([fast,slow] * fu_w^T + b) ---
    v8f ltc[4];
#pragma unroll
    for (int nt = 0; nt < 4; ++nt) {
      v8f acc = vzero8();
      acc = wmma_bf16(hAf, fuB[nt][0], acc);
      acc = wmma_bf16(hAs, fuB[nt][1], acc);
#pragma unroll
      for (int r = 0; r < 8; ++r) ltc[nt][r] = fast_tanh(acc[r] + fub[nt]);
    }
    // --- transpose ltc for attention ---
#pragma unroll
    for (int nt = 0; nt < 4; ++nt)
#pragma unroll
      for (int r = 0; r < 8; ++r)
        shB[(r + 8*h) * 68 + nt*16 + n0] = ltc[nt][r];
    __syncthreads();
    v16bf lA0 = afrag_lds(shB, 0,  lane);
    v16bf lA1 = afrag_lds(shB, 32, lane);

    // --- attention layer 1 (WMMA) + layer 2 (butterfly dot-32) ---
    v8f g0 = vzero8(), g1 = vzero8();
    g0 = wmma_bf16(lA0, at1B[0][0], g0);
    g0 = wmma_bf16(lA1, at1B[0][1], g0);
    g1 = wmma_bf16(lA0, at1B[1][0], g1);
    g1 = wmma_bf16(lA1, at1B[1][1], g1);
    float pr[8];
#pragma unroll
    for (int r = 0; r < 8; ++r)
      pr[r] = fast_tanh(g0[r] + ab1[0]) * w2a + fast_tanh(g1[r] + ab1[1]) * w2b;
#pragma unroll
    for (int r = 0; r < 8; ++r)
#pragma unroll
      for (int m = 1; m < 16; m <<= 1)
        pr[r] += __shfl_xor(pr[r], m, 16);

    // --- online softmax context accumulation ---
#pragma unroll
    for (int r = 0; r < 8; ++r) {
      float logit = pr[r] + ab2;
      float mn = fmaxf(mrow[r], logit);
      float al = fast_exp(mrow[r] - mn);
      float ee = fast_exp(logit - mn);
      srow[r] = srow[r] * al + ee;
#pragma unroll
      for (int nt = 0; nt < 4; ++nt)
        ctx[nt][r] = ctx[nt][r] * al + ee * ltc[nt][r];
      mrow[r] = mn;
    }

    ax = axn;
#pragma unroll
    for (int r = 0; r < 8; ++r) inv8[r] = invn[r];
  }

#pragma unroll
  for (int nt = 0; nt < 4; ++nt)
#pragma unroll
    for (int r = 0; r < 8; ++r)
      ctxg[(size_t)(tile*16 + r + 8*h) * 64 + nt*16 + n0] = ctx[nt][r] / srow[r];
}

// ---------------------------------------------------------------------------
// Kernel 3: output heads, one thread per batch row.
// ---------------------------------------------------------------------------
__global__ __launch_bounds__(256) void k3_heads(
    const float* __restrict__ ctxg,
    const float* __restrict__ dw1, const float* __restrict__ db1,
    const float* __restrict__ dw2, const float* __restrict__ db2,
    const float* __restrict__ ow1, const float* __restrict__ ob1,
    const float* __restrict__ ow2, const float* __restrict__ ob2,
    const float* __restrict__ ow3, const float* __restrict__ ob3,
    const float* __restrict__ iw1, const float* __restrict__ ib1,
    const float* __restrict__ iw2, const float* __restrict__ ib2,
    float* __restrict__ out) {
  int b = blockIdx.x * blockDim.x + threadIdx.x;
  if (b >= B_TOT) return;
  float c[64];
  for (int j = 0; j < 64; ++j) c[j] = ctxg[(size_t)b * 64 + j];

  float t1[64];
  for (int u = 0; u < 64; ++u) {
    float a = db1[u];
    for (int j = 0; j < 64; ++j) a += dw1[u*64 + j] * c[j];
    t1[u] = fmaxf(a, 0.f);
  }
  for (int u = 0; u < 5; ++u) {
    float a = db2[u];
    for (int j = 0; j < 64; ++j) a += dw2[u*64 + j] * t1[j];
    out[b * 5 + u] = a;
  }

  float t2[64];
  for (int u = 0; u < 64; ++u) {
    float a = ob1[u];
    for (int j = 0; j < 64; ++j) a += ow1[u*64 + j] * c[j];
    t2[u] = fmaxf(a, 0.f);
  }
  float t3[32];
  for (int u = 0; u < 32; ++u) {
    float a = ob2[u];
    for (int j = 0; j < 64; ++j) a += ow2[u*64 + j] * t2[j];
    t3[u] = fmaxf(a, 0.f);
  }
  float a3 = ob3[0];
  for (int j = 0; j < 32; ++j) a3 += ow3[j] * t3[j];
  out[5120 + b] = fast_sigmoid(a3);

  float t4[32];
  for (int u = 0; u < 32; ++u) {
    float a = ib1[u];
    for (int j = 0; j < 64; ++j) a += iw1[u*64 + j] * c[j];
    t4[u] = fmaxf(a, 0.f);
  }
  float a4 = ib2[0];
  for (int j = 0; j < 32; ++j) a4 += iw2[j] * t4[j];
  out[6144 + b] = fast_sigmoid(a4);
}

// ---------------------------------------------------------------------------
extern "C" void kernel_launch(void* const* d_in, const int* in_sizes, int n_in,
                              void* d_out, int out_size, void* d_ws, size_t ws_size,
                              hipStream_t stream) {
  (void)in_sizes; (void)n_in; (void)out_size; (void)ws_size;
  const float* x     = (const float*)d_in[0];
  const float* fg_w1 = (const float*)d_in[1];
  const float* fg_b1 = (const float*)d_in[2];
  const float* fg_w2 = (const float*)d_in[3];
  const float* fg_b2 = (const float*)d_in[4];
  const float* fwin  = (const float*)d_in[5];
  const float* fbin  = (const float*)d_in[6];
  const float* fwrec = (const float*)d_in[7];
  const float* swin  = (const float*)d_in[8];
  const float* sbin  = (const float*)d_in[9];
  const float* swrec = (const float*)d_in[10];
  const float* fu_w  = (const float*)d_in[11];
  const float* fu_b  = (const float*)d_in[12];
  const float* at_w1 = (const float*)d_in[13];
  const float* at_b1 = (const float*)d_in[14];
  const float* at_w2 = (const float*)d_in[15];
  const float* at_b2 = (const float*)d_in[16];
  const float* d_w1  = (const float*)d_in[17];
  const float* d_b1  = (const float*)d_in[18];
  const float* d_w2  = (const float*)d_in[19];
  const float* d_b2  = (const float*)d_in[20];
  const float* o_w1  = (const float*)d_in[21];
  const float* o_b1  = (const float*)d_in[22];
  const float* o_w2  = (const float*)d_in[23];
  const float* o_b2  = (const float*)d_in[24];
  const float* o_w3  = (const float*)d_in[25];
  const float* o_b3  = (const float*)d_in[26];
  const float* i_w1  = (const float*)d_in[27];
  const float* i_b1  = (const float*)d_in[28];
  const float* i_w2  = (const float*)d_in[29];
  const float* i_b2  = (const float*)d_in[30];

  char* ws = (char*)d_ws;
  unsigned short* fragx = (unsigned short*)ws;                        // 32 MB bf16 A-fragments
  float* invtau = (float*)(ws + (size_t)32 * 1024 * 1024);            // 2 MB
  float* ctxg   = (float*)(ws + (size_t)34 * 1024 * 1024);            // 256 KB

  k1_gate_repack<<<dim3(NTILE, T_LEN / TCH), 32, 0, stream>>>(
      x, fg_w1, fg_b1, fg_w2, fg_b2, fragx, invtau);

  k2_scan<<<dim3(NTILE), 32, 0, stream>>>(
      fragx, invtau, fwin, fbin, fwrec, swin, sbin, swrec,
      fu_w, fu_b, at_w1, at_b1, at_w2, at_b2, ctxg);

  k3_heads<<<dim3(4), 256, 0, stream>>>(
      ctxg, d_w1, d_b1, d_w2, d_b2, o_w1, o_b1, o_w2, o_b2,
      o_w3, o_b3, i_w1, i_b1, i_w2, i_b2, (float*)d_out);
}
